// TorchMoji_89773406421010
// MI455X (gfx1250) — matrix operand order repaired and checked by reading, not compiled, not run on hardware
//
#include <hip/hip_runtime.h>
#include <hip/hip_bf16.h>

typedef __bf16 bf16;
typedef __attribute__((ext_vector_type(16))) __bf16 v16bf;
typedef __attribute__((ext_vector_type(8)))  float  v8f;

constexpr int NB = 32;    // batch
constexpr int NT = 256;   // time
constexpr int NE = 256;   // embed
constexpr int NH = 512;   // hidden
constexpr int NG = 2048;  // gates = 4H
constexpr int NA = 2304;  // feats = 4H + E
constexpr int NC = 64;    // classes

__device__ __forceinline__ float hsig(float x) {
    return fminf(fmaxf(0.2f * x + 0.5f, 0.0f), 1.0f);
}

// Swizzle within each 32-wide K chunk so an A-fragment (16 bf16 per lane) is
// contiguous: positions [0..7]=K0..7, [8..15]=K16..23, [16..23]=K8..15, [24..31]=K24..31.
__device__ __forceinline__ int swz_pos(int off) {
    if (off >= 8 && off < 16)  return off + 8;
    if (off >= 16 && off < 24) return off - 8;
    return off;
}

// ---------------------------------------------------------------------------
// Embedding: x = tanh(embed[ids]) * mask  -> f32 (for feats) + swizzled bf16
// ---------------------------------------------------------------------------
__global__ __launch_bounds__(256) void embed_kernel(
    const int* __restrict__ ids, const int* __restrict__ lengths,
    const float* __restrict__ embed, float* __restrict__ x_f32,
    bf16* __restrict__ xsw)
{
    int b = blockIdx.x / NT, t = blockIdx.x % NT, e = threadIdx.x;
    bool m = t < lengths[b];
    float v = m ? tanhf(embed[(size_t)ids[b * NT + t] * NE + e]) : 0.0f;
    size_t row = (size_t)(b * NT + t);
    x_f32[row * NE + e] = v;
    int pos = (e & ~31) + swz_pos(e & 31);
    xsw[row * NE + pos] = (bf16)v;
}

// ---------------------------------------------------------------------------
// Pack a (2048 x Kin) f32 weight into WMMA B-fragment layout:
// dst[((nt*KTtot + ktbase + kt)*32 + lane)*16 + e] = w[nt*16 + (lane&15)][kt*32 + (e>>3)*16 + (lane>>4)*8 + (e&7)]
// ---------------------------------------------------------------------------
__global__ __launch_bounds__(256) void pack_b_kernel(
    const float* __restrict__ w, bf16* __restrict__ dst,
    int Kin, int KTtot, int ktbase)
{
    int idx = blockIdx.x * blockDim.x + threadIdx.x;
    int KTin = Kin >> 5;
    if (idx >= 128 * KTin * 32 * 16) return;
    int e = idx & 15; int r = idx >> 4;
    int lane = r & 31; r >>= 5;
    int kt = r % KTin; int nt = r / KTin;
    int n = nt * 16 + (lane & 15);
    int k = kt * 32 + (e >> 3) * 16 + (lane >> 4) * 8 + (e & 7);
    dst[(((size_t)nt * KTtot + ktbase + kt) * 32 + lane) * 16 + e] =
        (bf16)w[(size_t)n * Kin + k];
}

// ---------------------------------------------------------------------------
// LSTM scan (one direction per block). 1024 threads = 32 waves.
// Wave w owns h-columns [16w,16w+16): its 4 matching i/f/g/o gate tiles x
// 2 batch tiles. Per step: gates = [h | x_t] @ Wcat^T + b with
// v_wmma_f32_16x16x32_bf16. K loop is split into an h-phase (A from LDS,
// ds_load_b128) and an x-phase (A from global, global_load_b128) so each
// phase has a single address space. Cell state c stays in accumulator-layout
// VGPRs for all 256 steps.
// ---------------------------------------------------------------------------
__global__ __launch_bounds__(1024) void lstm_scan_kernel(
    const int* __restrict__ lengths,
    const bf16* __restrict__ xsw, int Kx,
    const bf16* __restrict__ Wp_f, const bf16* __restrict__ Wp_b,
    const float* __restrict__ bias_f, const float* __restrict__ bias_b,
    float* __restrict__ hout,    // (B,T,2H), this dir writes cols [dir*H, dir*H+H)
    bf16* __restrict__ hout_sw)  // (B,T,2H) swizzled bf16, or nullptr
{
    const int dir = blockIdx.x;
    const bf16*  Wp   = dir ? Wp_b : Wp_f;
    const float* bias = dir ? bias_b : bias_f;
    constexpr int KTH = NH >> 5;        // 16 k-tiles from h
    const int KTX = Kx >> 5;            // k-tiles from x
    const int KT  = KTH + KTX;

    __shared__ __align__(32) bf16 hbuf[NB * NH];  // swizzled h, 32 KB
    __shared__ int slen[NB];

    const int tid  = threadIdx.x;
    const int wave = tid >> 5;
    const int lane = tid & 31;
    const int lrow = lane & 15;
    const int half = lane >> 4;

    if (tid < NB) slen[tid] = lengths[tid];
    for (int i = tid; i < NB * NH; i += 1024) hbuf[i] = (bf16)0.0f;
    __syncthreads();

    const int hcol = wave * 16 + lrow;                  // this lane's h column
    const int hsw  = (hcol & ~31) + swz_pos(hcol & 31); // swizzled offset in 512-row

    const float bi  = bias[hcol];
    const float bfv = bias[NH + hcol];
    const float bg  = bias[2 * NH + hcol];
    const float bo  = bias[3 * NH + hcol];

    float cst[2][8];
    #pragma unroll
    for (int mt = 0; mt < 2; ++mt)
        #pragma unroll
        for (int r = 0; r < 8; ++r) cst[mt][r] = 0.0f;

    // per-lane B-fragment base pointers for gates i,f,g,o (nt = g*32 + wave)
    const bf16* wbase[4];
    #pragma unroll
    for (int g = 0; g < 4; ++g)
        wbase[g] = Wp + ((size_t)(g * 32 + wave) * KT * 32 + lane) * 16;

    // LDS A-fragment rows for the two batch tiles (pure shared-memory pointers)
    const bf16* hrow0 = &hbuf[(lrow     ) * NH + half * 16];
    const bf16* hrow1 = &hbuf[(lrow + 16) * NH + half * 16];

    for (int t = 0; t < NT; ++t) {
        v8f acc[2][4];
        #pragma unroll
        for (int mt = 0; mt < 2; ++mt)
            #pragma unroll
            for (int g = 0; g < 4; ++g) acc[mt][g] = {};

        // ---- h-phase: A from LDS (ds_load_b128), B from global weights ----
        #pragma unroll 4
        for (int kt = 0; kt < KTH; ++kt) {
            v16bf a0 = *(const v16bf*)(hrow0 + kt * 32);
            v16bf a1 = *(const v16bf*)(hrow1 + kt * 32);
            #pragma unroll
            for (int g = 0; g < 4; ++g) {
                v16bf bm = *(const v16bf*)(wbase[g] + (size_t)kt * 32 * 16);
                acc[0][g] = __builtin_amdgcn_wmma_f32_16x16x32_bf16(
                    false, a0, false, bm, (short)0, acc[0][g], false, false);
                acc[1][g] = __builtin_amdgcn_wmma_f32_16x16x32_bf16(
                    false, a1, false, bm, (short)0, acc[1][g], false, false);
            }
        }

        // ---- x-phase: A from global (global_load_b128), per-batch reversed time
        int tt0, tt1;
        {
            int L0 = slen[lrow];
            int L1 = slen[lrow + 16];
            tt0 = dir ? ((t < L0) ? (L0 - 1 - t) : t) : t;
            tt1 = dir ? ((t < L1) ? (L1 - 1 - t) : t) : t;
        }
        const bf16* xrow0 = xsw + ((size_t)(lrow     ) * NT + tt0) * Kx + half * 16;
        const bf16* xrow1 = xsw + ((size_t)(lrow + 16) * NT + tt1) * Kx + half * 16;
        #pragma unroll 4
        for (int kt = 0; kt < KTX; ++kt) {
            v16bf a0 = *(const v16bf*)(xrow0 + kt * 32);
            v16bf a1 = *(const v16bf*)(xrow1 + kt * 32);
            #pragma unroll
            for (int g = 0; g < 4; ++g) {
                v16bf bm = *(const v16bf*)(wbase[g] + (size_t)(KTH + kt) * 32 * 16);
                acc[0][g] = __builtin_amdgcn_wmma_f32_16x16x32_bf16(
                    false, a0, false, bm, (short)0, acc[0][g], false, false);
                acc[1][g] = __builtin_amdgcn_wmma_f32_16x16x32_bf16(
                    false, a1, false, bm, (short)0, acc[1][g], false, false);
            }
        }

        __syncthreads();  // all hbuf reads done before this step's writes

        #pragma unroll
        for (int mt = 0; mt < 2; ++mt) {
            union { v8f v; float f[8]; } ui, uf, ug, uo;
            ui.v = acc[mt][0]; uf.v = acc[mt][1];
            ug.v = acc[mt][2]; uo.v = acc[mt][3];
            #pragma unroll
            for (int r = 0; r < 8; ++r) {
                int batch = r + 8 * half + 16 * mt;  // C-layout row
                int L = slen[batch];
                bool m = t < L;
                float iv = hsig(ui.f[r] + bi);
                float fv = hsig(uf.f[r] + bfv);
                float gv = tanhf(ug.f[r] + bg);
                float ov = hsig(uo.f[r] + bo);
                float cn = fv * cst[mt][r] + iv * gv;
                float hn = ov * tanhf(cn);
                if (m) {                       // where(mask, new, old)
                    cst[mt][r] = cn;
                    hbuf[batch * NH + hsw] = (bf16)hn;
                }
                float outv = m ? hn : 0.0f;    // masked scan output
                int tout = dir ? (m ? (L - 1 - t) : t) : t;  // un-reverse bwd
                size_t rowo = ((size_t)batch * NT + tout) * (2 * NH) + dir * NH;
                hout[rowo + hcol] = outv;
                if (hout_sw) hout_sw[rowo + hsw] = (bf16)outv;
            }
        }
        __syncthreads();  // h writes visible before next step's fragment loads
    }
}

// ---------------------------------------------------------------------------
// Attention pooling + output head + softmax. One block per batch row.
// ---------------------------------------------------------------------------
__global__ __launch_bounds__(256) void attn_out_kernel(
    const int* __restrict__ lengths, const float* __restrict__ x,
    const float* __restrict__ h0, const float* __restrict__ h1,
    const float* __restrict__ att_v, const float* __restrict__ out_w,
    const float* __restrict__ out_b, float* __restrict__ out)
{
    __shared__ float satt[NT];
    __shared__ float rep[NA];
    __shared__ float red[256];
    __shared__ float olog[NC];
    int b = blockIdx.x, t = threadIdx.x;
    int L = lengths[b];
    bool m = t < L;
    size_t row = (size_t)b * NT + t;
    const float* h1r = h1 + row * (2 * NH);
    const float* h0r = h0 + row * (2 * NH);
    const float* xr  = x  + row * NE;

    float lg = 0.0f;
    for (int a = 0; a < 2 * NH; ++a) lg += h1r[a] * att_v[a];
    for (int a = 0; a < 2 * NH; ++a) lg += h0r[a] * att_v[2 * NH + a];
    for (int a = 0; a < NE; ++a)     lg += xr[a]  * att_v[4 * NH + a];

    red[t] = m ? lg : -1e30f;
    __syncthreads();
    for (int s = 128; s > 0; s >>= 1) { if (t < s) red[t] = fmaxf(red[t], red[t + s]); __syncthreads(); }
    float mx = red[0];
    __syncthreads();
    float e = m ? __expf(lg - mx) : 0.0f;
    satt[t] = e; red[t] = e;
    __syncthreads();
    for (int s = 128; s > 0; s >>= 1) { if (t < s) red[t] += red[t + s]; __syncthreads(); }
    float inv = 1.0f / red[0];
    __syncthreads();

    for (int a = t; a < NA; a += 256) {
        float s = 0.0f;
        for (int u = 0; u < NT; ++u) {
            size_t ru = (size_t)b * NT + u;
            float f = (a < 2 * NH)   ? h1[ru * (2 * NH) + a]
                    : (a < 4 * NH)   ? h0[ru * (2 * NH) + (a - 2 * NH)]
                                     : x[ru * NE + (a - 4 * NH)];
            s += satt[u] * f;
        }
        rep[a] = s * inv;
    }
    __syncthreads();

    if (t < NC) {
        float lo = out_b[t];
        for (int a = 0; a < NA; ++a) lo += rep[a] * out_w[(size_t)t * NA + a];
        olog[t] = lo;
    }
    __syncthreads();
    if (t < NC) {
        float mx2 = -1e30f;
        for (int j = 0; j < NC; ++j) mx2 = fmaxf(mx2, olog[j]);
        float s = 0.0f;
        for (int j = 0; j < NC; ++j) s += __expf(olog[j] - mx2);
        out[b * NC + t] = __expf(olog[t] - mx2) / s;
    }
}

// ---------------------------------------------------------------------------
extern "C" void kernel_launch(void* const* d_in, const int* in_sizes, int n_in,
                              void* d_out, int out_size, void* d_ws, size_t ws_size,
                              hipStream_t stream)
{
    const int*   ids     = (const int*)  d_in[0];
    const int*   lens    = (const int*)  d_in[1];
    const float* embed   = (const float*)d_in[2];
    const float* w_ih_0f = (const float*)d_in[3];
    const float* w_hh_0f = (const float*)d_in[4];
    const float* b_0f    = (const float*)d_in[5];
    const float* w_ih_0b = (const float*)d_in[6];
    const float* w_hh_0b = (const float*)d_in[7];
    const float* b_0b    = (const float*)d_in[8];
    const float* w_ih_1f = (const float*)d_in[9];
    const float* w_hh_1f = (const float*)d_in[10];
    const float* b_1f    = (const float*)d_in[11];
    const float* w_ih_1b = (const float*)d_in[12];
    const float* w_hh_1b = (const float*)d_in[13];
    const float* b_1b    = (const float*)d_in[14];
    const float* att_v   = (const float*)d_in[15];
    const float* out_w   = (const float*)d_in[16];
    const float* out_b   = (const float*)d_in[17];
    float* out = (float*)d_out;

    char* ws = (char*)d_ws;
    size_t off = 0;
    auto take = [&](size_t bytes) {
        char* p = ws + off;
        off = (off + bytes + 255) & ~(size_t)255;
        return p;
    };
    float* x_f32 = (float*)take((size_t)NB * NT * NE * 4);
    bf16*  xsw0  = (bf16*) take((size_t)NB * NT * NE * 2);
    float* h0f32 = (float*)take((size_t)NB * NT * 2 * NH * 4);
    bf16*  h0sw  = (bf16*) take((size_t)NB * NT * 2 * NH * 2);
    float* h1f32 = (float*)take((size_t)NB * NT * 2 * NH * 4);
    const int KT0 = (NH + NE) >> 5;       // 24 k-tiles (h:16 + x:8)
    const int KT1 = (NH + 2 * NH) >> 5;   // 48 k-tiles (h:16 + x:32)
    bf16* Wp0f = (bf16*)take((size_t)NG * (NH + NE) * 2);
    bf16* Wp0b = (bf16*)take((size_t)NG * (NH + NE) * 2);
    bf16* Wp1f = (bf16*)take((size_t)NG * (NH + 2 * NH) * 2);
    bf16* Wp1b = (bf16*)take((size_t)NG * (NH + 2 * NH) * 2);

    // 1. embedding -> x (f32 + swizzled bf16)
    embed_kernel<<<NB * NT, 256, 0, stream>>>(ids, lens, embed, x_f32, xsw0);

    // 2. pack weights: [W_hh (kt 0..15) | W_ih (kt 16..)] per direction
    int nHH = NG * NH / 256, nIH0 = NG * NE / 256, nIH1 = NG * 2 * NH / 256;
    pack_b_kernel<<<nHH, 256, 0, stream>>>(w_hh_0f, Wp0f, NH,     KT0, 0);
    pack_b_kernel<<<nIH0, 256, 0, stream>>>(w_ih_0f, Wp0f, NE,     KT0, 16);
    pack_b_kernel<<<nHH, 256, 0, stream>>>(w_hh_0b, Wp0b, NH,     KT0, 0);
    pack_b_kernel<<<nIH0, 256, 0, stream>>>(w_ih_0b, Wp0b, NE,     KT0, 16);
    pack_b_kernel<<<nHH, 256, 0, stream>>>(w_hh_1f, Wp1f, NH,     KT1, 0);
    pack_b_kernel<<<nIH1, 256, 0, stream>>>(w_ih_1f, Wp1f, 2 * NH, KT1, 16);
    pack_b_kernel<<<nHH, 256, 0, stream>>>(w_hh_1b, Wp1b, NH,     KT1, 0);
    pack_b_kernel<<<nIH1, 256, 0, stream>>>(w_ih_1b, Wp1b, 2 * NH, KT1, 16);

    // 3. layer 0 bilstm (fwd+bwd concurrently, one block each)
    lstm_scan_kernel<<<2, 1024, 0, stream>>>(
        lens, xsw0, NE, Wp0f, Wp0b, b_0f, b_0b, h0f32, h0sw);

    // 4. layer 1 bilstm (input = swizzled h0)
    lstm_scan_kernel<<<2, 1024, 0, stream>>>(
        lens, h0sw, 2 * NH, Wp1f, Wp1b, b_1f, b_1b, h1f32, (bf16*)nullptr);

    // 5. attention pooling + output head + softmax
    attn_out_kernel<<<NB, 256, 0, stream>>>(
        lens, x_f32, h0f32, h1f32, att_v, out_w, out_b, out);
}
